// KMeansCriterion_2138893713651
// MI455X (gfx1250) — matrix operand order repaired and checked by reading, not compiled, not run on hardware
//
#include <hip/hip_runtime.h>
#include <hip/hip_bf16.h>

// ---------------------------------------------------------------------------
// KMeans criterion for MI455X (gfx1250, wave32, WMMA).
//   distances[n,k] = L*(|e_n|^2 + |c_k|^2 - 2 e_n.c_k)
//   m[n]   = max_k distances[n,k]       (argmax: first max)
//   loss   = L * sum_n m[n]
// Cross term via v_wmma_f32_16x16x32_bf16 (fp32 accumulate); |e|^2,|c|^2 exact fp32.
// Round 2: dual accumulator chains (2 column tiles / iter) to hide WMMA RAW
// latency and B-load waits, and to fill the WMMA->VALU hazard slots.
// ---------------------------------------------------------------------------

typedef __attribute__((ext_vector_type(16))) __bf16 v16bf;
typedef __attribute__((ext_vector_type(8)))  float  v8f;

#define DIMS   256      // D
#define NCENT  512      // K
#define ROWS_PER_BLK 64 // embedding rows per workgroup (4 waves x 16 rows)
#define LMBDA  0.25f

union ABFrag { uint4 u[2]; v16bf v; };
struct __align__(8) us4 { unsigned short x, y, z, w; };

__device__ __forceinline__ unsigned short f2bf(float f) {
    unsigned u = __float_as_uint(f);
    unsigned r = u + 0x7FFFu + ((u >> 16) & 1u);   // round-to-nearest-even
    return (unsigned short)(r >> 16);
}

// ---------------------------------------------------------------------------
// Kernel 1: centroids fp32 -> bf16 (workspace) + exact c_sq[k]
// grid = 512 blocks (one per centroid), block = 256 threads (one per dim)
// ---------------------------------------------------------------------------
__global__ void kmeans_prep_centroids(const float* __restrict__ C,
                                      unsigned short* __restrict__ cbf,
                                      float* __restrict__ c_sq) {
    const int k = blockIdx.x;
    const int t = threadIdx.x;
    float v = C[(size_t)k * DIMS + t];
    cbf[(size_t)k * DIMS + t] = f2bf(v);

    __shared__ float red[256];
    red[t] = v * v;
    __syncthreads();
    for (int s = 128; s > 0; s >>= 1) {
        if (t < s) red[t] += red[t + s];
        __syncthreads();
    }
    if (t == 0) c_sq[k] = red[0];
}

// ---------------------------------------------------------------------------
// Kernel 2: main WMMA distance + row max/argmax
// grid = N/64 blocks, block = 128 threads = 4 waves, each wave owns 16 rows
// ---------------------------------------------------------------------------
__global__ void kmeans_main(const float* __restrict__ E,
                            const unsigned short* __restrict__ cbf,
                            const float* __restrict__ c_sq,
                            float* __restrict__ out,        // [0]=loss (later), [1..N]=assignments
                            float* __restrict__ cdist) {    // [N] cluster_distances
    const int tid  = threadIdx.x;
    const int lane = tid & 31;
    const int w    = tid >> 5;          // wave id 0..3
    const int row0 = blockIdx.x * ROWS_PER_BLK;

    __shared__ __align__(16) unsigned short sE[ROWS_PER_BLK * DIMS]; // bf16 tile, 32 KB
    __shared__ float sEsq[ROWS_PER_BLK];

    // ---- stage: fp32 global -> bf16 LDS, exact |e|^2 per row ----
    for (int i = 0; i < ROWS_PER_BLK / 4; ++i) {
        const int lr = i * 4 + w;                       // local row this wave fills
        const float* src = E + (size_t)(row0 + lr) * DIMS;
        float ss = 0.f;
        #pragma unroll
        for (int j = 0; j < 2; ++j) {
            const int col = 4 * lane + 128 * j;         // 512B coalesced per wave
            float4 v = *(const float4*)(src + col);
            ss += v.x * v.x + v.y * v.y + v.z * v.z + v.w * v.w;
            us4 hh; hh.x = f2bf(v.x); hh.y = f2bf(v.y); hh.z = f2bf(v.z); hh.w = f2bf(v.w);
            *(us4*)&sE[lr * DIMS + col] = hh;
        }
        #pragma unroll
        for (int off = 16; off >= 1; off >>= 1) ss += __shfl_xor(ss, off, 32);
        if (lane == 0) sEsq[lr] = ss;
    }
    __syncthreads();

    // ---- preload A fragments (8 K-slices of 16x32 bf16) from LDS ----
    // ISA A layout: lanes 0-15 hold row M=lane, K {0..7,16..23}; lanes 16-31 row M=lane-16, K {8..15,24..31}
    const int h   = lane >> 4;          // half-wave select
    const int l16 = lane & 15;
    const int aRow = 16 * w + l16;      // local LDS row this lane supplies
    v16bf a[8];
    #pragma unroll
    for (int dt = 0; dt < 8; ++dt) {
        const unsigned short* p = &sE[aRow * DIMS + dt * 32 + h * 8];
        ABFrag f;
        f.u[0] = *(const uint4*)(p);        // K = dt*32 + h*8 .. +7
        f.u[1] = *(const uint4*)(p + 16);   // K = dt*32 + 16 + h*8 .. +7
        a[dt] = f.v;
    }

    float runmax[8];
    int   runidx[8];
    #pragma unroll
    for (int r = 0; r < 8; ++r) { runmax[r] = -3.402823e38f; runidx[r] = 0; }

    // ---- sweep 32 centroid tiles, two tiles (two acc chains) per iteration ----
    for (int ct = 0; ct < 32; ct += 2) {
        v8f acc0 = {};
        v8f acc1 = {};
        // B layout (matches A): lane holds column N=l16; lanes 0-15 K {0..7,16..23}, 16-31 K {8..15,24..31}
        const unsigned short* bp0 = cbf + (size_t)(ct * 16 + l16) * DIMS + h * 8;
        const unsigned short* bp1 = bp0 + (size_t)16 * DIMS;
        #pragma unroll
        for (int dt = 0; dt < 8; ++dt) {
            ABFrag b0, b1;
            b0.u[0] = *(const uint4*)(bp0 + dt * 32);      // K = dt*32 + h*8 .. +7
            b0.u[1] = *(const uint4*)(bp0 + dt * 32 + 16); // K = dt*32 + 16 + h*8 .. +7
            b1.u[0] = *(const uint4*)(bp1 + dt * 32);
            b1.u[1] = *(const uint4*)(bp1 + dt * 32 + 16);
            // Two independent chains: each WMMA's accumulator RAW is covered by
            // the other chain's issue; B loads overlap the opposite chain.
            acc0 = __builtin_amdgcn_wmma_f32_16x16x32_bf16(
                false, a[dt], false, b0.v, (short)0, acc0, false, false);
            acc1 = __builtin_amdgcn_wmma_f32_16x16x32_bf16(
                false, a[dt], false, b1.v, (short)0, acc1, false, false);
        }
        const int n0 = ct * 16 + l16;
        const int n1 = n0 + 16;
        const float cs0 = c_sq[n0];
        const float cs1 = c_sq[n1];
        #pragma unroll
        for (int r = 0; r < 8; ++r) {
            float v0 = cs0 - 2.0f * acc0[r];            // e_sq added later (row-constant)
            float v1 = cs1 - 2.0f * acc1[r];
            if (v0 > runmax[r]) { runmax[r] = v0; runidx[r] = n0; }
            if (v1 > runmax[r]) { runmax[r] = v1; runidx[r] = n1; }
        }
    }

    // ---- reduce max/argmax across the 16 lanes sharing each row ----
    #pragma unroll
    for (int r = 0; r < 8; ++r) {
        float m  = runmax[r];
        int   ix = runidx[r];
        #pragma unroll
        for (int mask = 1; mask < 16; mask <<= 1) {
            float om = __shfl_xor(m, mask, 32);
            int   oi = __shfl_xor(ix, mask, 32);
            if (om > m || (om == m && oi < ix)) { m = om; ix = oi; } // first-max tiebreak
        }
        if (l16 == 0) {
            const int lr = 16 * w + r + 8 * h;          // C-layout row: M = r + 8*half
            const int gr = row0 + lr;
            out[1 + gr]  = (float)ix;                   // assignments (float buffer)
            cdist[gr]    = LMBDA * (sEsq[lr] + m);      // cluster_distances[n]
        }
    }
}

// ---------------------------------------------------------------------------
// Kernel 3a/3b: deterministic two-stage sum -> loss
// ---------------------------------------------------------------------------
__global__ void kmeans_reduce1(const float* __restrict__ cdist,
                               float* __restrict__ partial) {
    __shared__ float red[256];
    const int t = threadIdx.x;
    red[t] = cdist[(size_t)blockIdx.x * 256 + t];
    __syncthreads();
    for (int s = 128; s > 0; s >>= 1) {
        if (t < s) red[t] += red[t + s];
        __syncthreads();
    }
    if (t == 0) partial[blockIdx.x] = red[0];
}

__global__ void kmeans_reduce2(const float* __restrict__ partial,
                               float* __restrict__ out) {
    __shared__ float red[256];
    const int t = threadIdx.x;
    red[t] = partial[t] + partial[t + 256];
    __syncthreads();
    for (int s = 128; s > 0; s >>= 1) {
        if (t < s) red[t] += red[t + s];
        __syncthreads();
    }
    if (t == 0) out[0] = LMBDA * red[0];   // loss = lmbda * sum(cluster_distances)
}

// ---------------------------------------------------------------------------
extern "C" void kernel_launch(void* const* d_in, const int* in_sizes, int n_in,
                              void* d_out, int out_size, void* d_ws, size_t ws_size,
                              hipStream_t stream) {
    const float* E = (const float*)d_in[0];   // [131072, 256] fp32
    const float* C = (const float*)d_in[1];   // [512, 256]    fp32
    float* out = (float*)d_out;               // [0]=loss, [1..131072]=assignments

    const int N = in_sizes[0] / DIMS;         // 131072
    const int K = in_sizes[1] / DIMS;         // 512

    // workspace layout
    char* ws = (char*)d_ws;
    unsigned short* cbf  = (unsigned short*)(ws);                       // 512*256*2 = 262144 B
    float*          c_sq = (float*)(ws + 262144);                       // 2048 B
    float*          cdist = (float*)(ws + 262144 + 2048);               // 131072*4 = 524288 B
    float*          partial = (float*)(ws + 262144 + 2048 + 524288);    // 2048 B

    kmeans_prep_centroids<<<K, 256, 0, stream>>>(C, cbf, c_sq);
    kmeans_main<<<N / ROWS_PER_BLK, 128, 0, stream>>>(E, cbf, c_sq, out, cdist);
    kmeans_reduce1<<<N / 256, 256, 0, stream>>>(cdist, partial);
    kmeans_reduce2<<<1, 256, 0, stream>>>(partial, out);
}
